// DIGNN_80642305950136
// MI455X (gfx1250) — matrix operand order repaired and checked by reading, not compiled, not run on hardware
//
#include <hip/hip_runtime.h>

#define NN 200000
#define EE 400000
#define GG 4096
#define EMB 300
#define FEAT 256
#define LAYERS 5
#define NUM_BOND 5
#define NUM_DIR 3
#define KPAD_MAX 320

typedef __attribute__((ext_vector_type(16))) __bf16 bf16x16;
typedef __attribute__((ext_vector_type(8)))  float  f32x8;

static __device__ __forceinline__ unsigned short f2bf(float f) {
    unsigned int u = __float_as_uint(f);
    unsigned int r = (u + 0x7FFFu + ((u >> 16) & 1u)) >> 16;
    return (unsigned short)r;
}

// ---------------- elementwise / setup kernels ----------------

__global__ void fill_f32(float* p, float v, int n) {
    for (int i = blockIdx.x * blockDim.x + threadIdx.x; i < n; i += gridDim.x * blockDim.x)
        p[i] = v;
}

__global__ void embed_k(const int* __restrict__ x, const float* __restrict__ emb1,
                        const float* __restrict__ emb2, float* __restrict__ h) {
    const int total = NN * EMB;
    for (int i = blockIdx.x * blockDim.x + threadIdx.x; i < total; i += gridDim.x * blockDim.x) {
        int node = i / EMB, c = i - node * EMB;
        h[i] = emb1[x[node * 2 + 0] * EMB + c] + emb2[x[node * 2 + 1] * EMB + c];
    }
}

// f32 [Kt x Nc] row-major -> bf16 [Np x Kp], column-k-major, zero padded
__global__ void conv_w_k(const float* __restrict__ src, unsigned short* __restrict__ dst,
                         int Kt, int Nc, int Kp, int Np) {
    const int total = Np * Kp;
    for (int i = blockIdx.x * blockDim.x + threadIdx.x; i < total; i += gridDim.x * blockDim.x) {
        int n = i / Kp, k = i - n * Kp;
        float v = (k < Kt && n < Nc) ? src[k * Nc + n] : 0.0f;
        dst[n * Kp + k] = f2bf(v);
    }
}

// ---------------- WMMA bf16 GEMM ----------------
// C[M x Nc] = A[M x Kt] (f32, lda=Kt) * B (bf16 [TJ*64 x Kp] column-k-major) [+ bias][relu]
// One block per 16-row tile, full output width: 4 waves, wave w owns column tiles
// {w*16 + 64*j : j<TJ}. A fragment loaded once per k-step, reused across TJ WMMAs.
template <int TJ>
__global__ void __launch_bounds__(128)
gemm_bf16_wmma(const float* __restrict__ A, const unsigned short* __restrict__ B,
               float* __restrict__ C, const float* __restrict__ bias,
               int Kt, int Kp, int Nc, int ldc, int relu) {
    __shared__ __align__(16) unsigned short sA[16 * KPAD_MAX];
    const int tid = threadIdx.x;
    const int rowBase = blockIdx.x * 16;

    // stage + convert A tile (16 x Kp) to bf16 in LDS (read h exactly once)
    for (int idx = tid; idx < 16 * Kp; idx += 128) {
        int r = idx / Kp, k = idx - r * Kp;
        float v = (k < Kt) ? A[(long)(rowBase + r) * Kt + k] : 0.0f;
        sA[r * Kp + k] = f2bf(v);
    }
    __syncthreads();

    const int wave = tid >> 5, lane = tid & 31;
    const int m = lane & 15;                 // A row / B column within tile
    const int aHalf = (lane < 16) ? 0 : 16;  // A: lanes 16-31 hold K=16..31
    const int bHalf = (lane < 16) ? 0 : 16;  // B: lanes 16-31 hold K=16..31

    union U { bf16x16 v; unsigned int u[8]; };
    U au, bu[TJ];
    f32x8 acc[TJ];
#pragma unroll
    for (int j = 0; j < TJ; ++j) acc[j] = (f32x8){0.f, 0.f, 0.f, 0.f, 0.f, 0.f, 0.f, 0.f};

    const unsigned short* Bbase = B + (long)(wave * 16 + m) * Kp;

    for (int ks = 0; ks < Kp; ks += 32) {
#pragma unroll
        for (int g = 0; g < 8; ++g) {
            int ka = 2 * g + aHalf;
            au.u[g] = *(const unsigned int*)&sA[m * Kp + ks + ka];
        }
#pragma unroll
        for (int j = 0; j < TJ; ++j) {
            const unsigned short* Bc = Bbase + (long)(64 * j) * Kp + ks + bHalf;
#pragma unroll
            for (int g = 0; g < 8; ++g)
                bu[j].u[g] = *(const unsigned int*)&Bc[2 * g];
        }
#pragma unroll
        for (int j = 0; j < TJ; ++j)
            acc[j] = __builtin_amdgcn_wmma_f32_16x16x32_bf16(
                false, au.v, false, bu[j].v, (short)0, acc[j], false, false);
    }

    const int mBase = (lane >> 4) * 8;  // C: lanes 0-15 -> M=VGPR, lanes 16-31 -> M=VGPR+8
#pragma unroll
    for (int j = 0; j < TJ; ++j) {
        const int col = wave * 16 + 64 * j + m;
        if (col < Nc) {
            const float bv = bias ? bias[col] : 0.0f;
#pragma unroll
            for (int r = 0; r < 8; ++r) {
                float v = acc[j][r] + bv;
                if (relu) v = fmaxf(v, 0.0f);
                C[(long)(rowBase + mBase + r) * ldc + col] = v;
            }
        }
    }
}

// ---------------- message scatter (segment_sum over dst), 4 channels/thread ----------------
__global__ void scatter_k(const float* __restrict__ hw, float* __restrict__ hdst,
                          const int* __restrict__ ei, const int* __restrict__ ea,
                          const float* __restrict__ e1, const float* __restrict__ e2) {
    const int C4 = EMB / 4;
    const int total = (EE + NN) * C4;
    for (int i = blockIdx.x * blockDim.x + threadIdx.x; i < total; i += gridDim.x * blockDim.x) {
        int e = i / C4, c = (i - e * C4) * 4;
        int s, d; float eb;
        if (e < EE) {
            s = ei[e]; d = ei[EE + e];
            eb = e1[ea[e * 2 + 0]] + e2[ea[e * 2 + 1]];
        } else {                       // self loop: type 4, dir 0
            s = d = e - EE;
            eb = e1[4] + e2[0];
        }
        const float4 v = *(const float4*)&hw[s * EMB + c];
        float* dp = &hdst[d * EMB + c];
        atomicAdd(dp + 0, v.x + eb);
        atomicAdd(dp + 1, v.y + eb);
        atomicAdd(dp + 2, v.z + eb);
        atomicAdd(dp + 3, v.w + eb);
    }
}

// ---------------- batchnorm ----------------
#define BN_ROWS 512
__global__ void bn_stats_k(const float* __restrict__ h, float* __restrict__ stats) {
    const int c = threadIdx.x;           // blockDim = 320
    if (c >= EMB) return;
    const int r0 = blockIdx.x * BN_ROWS;
    float s = 0.f, s2 = 0.f;
    for (int i = 0; i < BN_ROWS; ++i) {
        int r = r0 + i;
        if (r < NN) { float v = h[r * EMB + c]; s += v; s2 += v * v; }
    }
    atomicAdd(&stats[c], s);
    atomicAdd(&stats[EMB + c], s2);
}

__global__ void bn_apply_k(float* __restrict__ h, const float* __restrict__ stats,
                           const float* __restrict__ gamma, const float* __restrict__ beta,
                           int relu) {
    const int total = NN * EMB;
    const float invN = 1.0f / (float)NN;
    for (int i = blockIdx.x * blockDim.x + threadIdx.x; i < total; i += gridDim.x * blockDim.x) {
        int c = i % EMB;
        float mean = stats[c] * invN;
        float var = stats[EMB + c] * invN - mean * mean;
        float v = (h[i] - mean) * rsqrtf(var + 1e-5f) * gamma[c] + beta[c];
        if (relu) v = fmaxf(v, 0.0f);
        h[i] = v;
    }
}

// ---------------- global mean pool ----------------
__global__ void pool_sum_k(const float* __restrict__ h, const int* __restrict__ batch,
                           float* __restrict__ hg) {
    const int C4 = EMB / 4;
    const int total = NN * C4;
    for (int i = blockIdx.x * blockDim.x + threadIdx.x; i < total; i += gridDim.x * blockDim.x) {
        int node = i / C4, c = (i - node * C4) * 4;
        const float4 v = *(const float4*)&h[node * EMB + c];
        float* gp = &hg[batch[node] * EMB + c];
        atomicAdd(gp + 0, v.x);
        atomicAdd(gp + 1, v.y);
        atomicAdd(gp + 2, v.z);
        atomicAdd(gp + 3, v.w);
    }
}

__global__ void pool_cnt_k(const int* __restrict__ batch, float* __restrict__ cnt) {
    for (int i = blockIdx.x * blockDim.x + threadIdx.x; i < NN; i += gridDim.x * blockDim.x)
        atomicAdd(&cnt[batch[i]], 1.0f);
}

__global__ void pool_div_k(float* __restrict__ hg, const float* __restrict__ cnt) {
    const int total = GG * EMB;
    for (int i = blockIdx.x * blockDim.x + threadIdx.x; i < total; i += gridDim.x * blockDim.x)
        hg[i] /= fmaxf(cnt[i / EMB], 1.0f);
}

// ---------------- host launcher ----------------
static inline int nblk(long total, int bs) {
    long b = (total + bs - 1) / bs;
    return (int)(b > 131072 ? 131072 : b);
}

extern "C" void kernel_launch(void* const* d_in, const int* in_sizes, int n_in,
                              void* d_out, int out_size, void* d_ws, size_t ws_size,
                              hipStream_t stream) {
    (void)in_sizes; (void)n_in; (void)out_size; (void)ws_size;
    const int*   x      = (const int*)d_in[0];
    const int*   ei     = (const int*)d_in[1];   // [2, E]
    const int*   ea     = (const int*)d_in[2];   // [E, 2]
    const int*   batch  = (const int*)d_in[3];
    const float* emb1   = (const float*)d_in[4];
    const float* emb2   = (const float*)d_in[5];
    const float* W      = (const float*)d_in[6];   // [L,300,300]
    // d_in[7] = b[l] : cancels exactly through batchnorm mean subtraction -> skipped
    const float* e1     = (const float*)d_in[8];   // [L,5,1]
    const float* e2     = (const float*)d_in[9];   // [L,3,1]
    const float* gamma  = (const float*)d_in[10];
    const float* beta   = (const float*)d_in[11];
    const float* featW  = (const float*)d_in[12];  // [300,256]
    const float* featB  = (const float*)d_in[13];
    const float* outW1  = (const float*)d_in[14];  // [256,256]
    const float* outB1  = (const float*)d_in[15];
    const float* outW2  = (const float*)d_in[16];  // [256,128]
    const float* outB2  = (const float*)d_in[17];
    float* out = (float*)d_out;                    // hf [4096*256] ++ out [4096*128]

    char* w = (char*)d_ws;
    float* hA = (float*)w;            w += (size_t)NN * EMB * 4;       // current h
    float* hB = (float*)w;            w += (size_t)NN * EMB * 4;       // hw scratch
    unsigned short* Wbf = (unsigned short*)w; w += (size_t)KPAD_MAX * KPAD_MAX * 2;
    float* stats = (float*)w;         w += 4096;                       // 600 used
    float* hg = (float*)w;            w += (size_t)GG * EMB * 4;       // also reused as t1
    float* cnt = (float*)w;           w += (size_t)GG * 4;
    float* t1 = hg;                                                    // [4096,256] after pooling done

    const int BS = 256;

    // h = emb1[x0] + emb2[x1]
    embed_k<<<nblk((long)NN * EMB, BS), BS, 0, stream>>>(x, emb1, emb2, hA);

    for (int l = 0; l < LAYERS; ++l) {
        // W[l] -> bf16 padded 320x320, column-k-major
        conv_w_k<<<nblk(320 * 320, BS), BS, 0, stream>>>(
            W + (size_t)l * EMB * EMB, Wbf, EMB, EMB, 320, 320);
        // hw = h @ W[l]  (full width per block: 4 waves x 5 col tiles)
        gemm_bf16_wmma<5><<<NN / 16, 128, 0, stream>>>(
            hA, Wbf, hB, nullptr, EMB, 320, EMB, EMB, 0);
        // h_new = segment_sum(hw[src] + edge_emb, dst)   (b[l] cancels in BN)
        fill_f32<<<nblk((long)NN * EMB, BS), BS, 0, stream>>>(hA, 0.0f, NN * EMB);
        scatter_k<<<nblk((long)(EE + NN) * (EMB / 4), BS), BS, 0, stream>>>(
            hB, hA, ei, ea, e1 + l * NUM_BOND, e2 + l * NUM_DIR);
        // batchnorm (+ relu except last layer)
        fill_f32<<<1, 640, 0, stream>>>(stats, 0.0f, 2 * EMB);
        bn_stats_k<<<(NN + BN_ROWS - 1) / BN_ROWS, 320, 0, stream>>>(hA, stats);
        bn_apply_k<<<nblk((long)NN * EMB, BS), BS, 0, stream>>>(
            hA, stats, gamma + l * EMB, beta + l * EMB, (l < LAYERS - 1) ? 1 : 0);
    }

    // global mean pool
    fill_f32<<<nblk((long)GG * EMB, BS), BS, 0, stream>>>(hg, 0.0f, GG * EMB);
    fill_f32<<<nblk(GG, BS), BS, 0, stream>>>(cnt, 0.0f, GG);
    pool_sum_k<<<nblk((long)NN * (EMB / 4), BS), BS, 0, stream>>>(hA, batch, hg);
    pool_cnt_k<<<nblk(NN, BS), BS, 0, stream>>>(batch, cnt);
    pool_div_k<<<nblk((long)GG * EMB, BS), BS, 0, stream>>>(hg, cnt);

    // hf = hg @ feat_W + feat_b        -> d_out[0 : 4096*256]
    conv_w_k<<<nblk(256 * 320, BS), BS, 0, stream>>>(featW, Wbf, EMB, FEAT, 320, 256);
    gemm_bf16_wmma<4><<<GG / 16, 128, 0, stream>>>(
        hg, Wbf, out, featB, EMB, 320, FEAT, FEAT, 0);

    // t1 = relu(hf @ out_W1 + out_b1)  (t1 reuses hg buffer; hg no longer needed)
    conv_w_k<<<nblk(256 * 256, BS), BS, 0, stream>>>(outW1, Wbf, FEAT, FEAT, 256, 256);
    gemm_bf16_wmma<4><<<GG / 16, 128, 0, stream>>>(
        out, Wbf, t1, outB1, FEAT, 256, FEAT, FEAT, 1);

    // out = t1 @ out_W2 + out_b2       -> d_out[4096*256 : ]
    conv_w_k<<<nblk(128 * 256, BS), BS, 0, stream>>>(outW2, Wbf, FEAT, FEAT / 2, 256, 128);
    gemm_bf16_wmma<2><<<GG / 16, 128, 0, stream>>>(
        t1, Wbf, out + (size_t)GG * FEAT, outB2, FEAT, 256, FEAT / 2, FEAT / 2, 0);
}
